// MNPNDMambaEncoder_42279658062355
// MI455X (gfx1250) — matrix operand order repaired and checked by reading, not compiled, not run on hardware
//
#include <hip/hip_runtime.h>
#include <hip/hip_bf16.h>
#include <math.h>

// ---------------------------------------------------------------------------
// Mamba encoder for MI455X (gfx1250, wave32, WMMA).
// GEMMs: v_wmma_f32_16x16x32_bf16 with fragment-major LDS staging so each
// lane's operand fragment is two ds_load_b128 (no scalar u16 gathers).
// ---------------------------------------------------------------------------

#define NUM_LAYERS 4
#define D_MODEL    512
#define D_INNER    1024
#define D_STATE    16
#define DT_RANK    32
#define D_CONV     4
#define BATCH      2
#define NCTX       768
#define NTGT       256
#define LSEQ       1024            // NCTX + NTGT
#define NTOK       (BATCH * LSEQ)  // 2048

typedef __attribute__((ext_vector_type(16))) __bf16        v16bf;
typedef __attribute__((ext_vector_type(8)))  float         v8f;
typedef __attribute__((ext_vector_type(4)))  unsigned int  u32x4;

// ------------------------------ WMMA GEMM ----------------------------------
// C[M,N] = epilogue( A[M,K](lda) * B[K,N](ldb) )
// Block: 128 threads (4 waves). Block tile 128(M) x 64(N); wave tile 32x64
// (2 A-frags x 4 B-frags = 8 wmma per 32-deep K chunk).
#define TILE_M 128
#define TILE_N 64
#define TILE_K 32
#define LDS_STRIDE 40   // 32 bf16 payload + pad; 80 bytes => 16B-aligned rows

#define EPI_STORE    0
#define EPI_SOFTPLUS 1   // C = softplus(acc + bias[col])
#define EPI_RESID    2   // C = acc + resid[row*ldc+col]

// LDS slot permutation for the A operand (ISA 7.12.2 16-bit A 16x32 layout):
// fragment element i of lane-half h maps to K = (i>>3)*16 + h*8 + (i&7);
// inverse: K -> slot = ((k>>3)&1)*16 + (k>>4)*8 + (k&7). Even-k pairs stay
// adjacent, so staging can use packed 2xbf16 dword stores.
__device__ __forceinline__ int a_slot(int k) {
    return (((k >> 3) & 1) << 4) + ((k >> 4) << 3) + (k & 7);
}

__device__ __forceinline__ void store_bf16x2(__bf16* p, float a, float b) {
    union { unsigned int u; __bf16 h[2]; } pk;
    pk.h[0] = (__bf16)a; pk.h[1] = (__bf16)b;
    *(unsigned int*)p = pk.u;            // 4B-aligned (even slot)
}

__device__ __forceinline__ v16bf load_frag(const __bf16* p) {
    // p is 16B aligned: two ds_load_b128
    union { u32x4 q[2]; v16bf v; } f;
    f.q[0] = *(const u32x4*)p;
    f.q[1] = *(const u32x4*)(p + 8);
    return f.v;
}

__global__ __launch_bounds__(128)
void wmma_gemm_kernel(const float* __restrict__ A, int lda,
                      const float* __restrict__ B, int ldb,
                      float* __restrict__ C, int ldc,
                      int M, int N, int K,
                      const float* __restrict__ bias,
                      const float* __restrict__ resid,
                      int epi)
{
    __shared__ __bf16 sA[TILE_M][LDS_STRIDE];   // permuted fragment-major rows
    __shared__ __bf16 sBt[TILE_N][LDS_STRIDE];  // B transposed: sBt[n][k]

    const int m0   = blockIdx.y * TILE_M;
    const int n0   = blockIdx.x * TILE_N;
    const int tid  = threadIdx.x;
    const int wave = tid >> 5;
    const int lane = tid & 31;
    const int hl   = lane >> 4;      // lane half (ISA lane groups)
    const int l16  = lane & 15;

    v8f acc[2][4] = {};              // 2 M-strips x 4 N-tiles per wave

    for (int k0 = 0; k0 < K; k0 += TILE_K) {
        // --- A: 128x32 floats, read float2 coalesced over K, store packed ---
        #pragma unroll
        for (int j = 0; j < 16; ++j) {
            const int p  = tid + 128 * j;        // pair index (4096/2)
            const int r  = p >> 4;               // row in tile
            const int k  = (p & 15) * 2;         // even K
            const float2 a2 = *(const float2*)&A[(size_t)(m0 + r) * lda + (k0 + k)];
            store_bf16x2(&sA[r][a_slot(k)], a2.x, a2.y);
        }
        // --- B: 32x64 floats, read coalesced over N (2 k-rows/thread),
        //        store packed into the transpose ---
        #pragma unroll
        for (int j = 0; j < 8; ++j) {
            const int pr = tid + 128 * j;        // 1024 (n, k-pair) tasks
            const int n  = pr & 63;
            const int k  = (pr >> 6) * 2;
            const float b0 = B[(size_t)(k0 + k)     * ldb + (n0 + n)];
            const float b1 = B[(size_t)(k0 + k + 1) * ldb + (n0 + n)];
            store_bf16x2(&sBt[n][a_slot(k)], b0, b1);
        }
        __syncthreads();

        const v16bf af0 = load_frag(&sA[32 * wave + l16     ][hl * 16]);
        const v16bf af1 = load_frag(&sA[32 * wave + 16 + l16][hl * 16]);
        #pragma unroll
        for (int t = 0; t < 4; ++t) {
            const v16bf bf = load_frag(&sBt[16 * t + l16][hl * 16]);
            acc[0][t] = __builtin_amdgcn_wmma_f32_16x16x32_bf16(
                false, af0, false, bf, (short)0, acc[0][t], false, false);
            acc[1][t] = __builtin_amdgcn_wmma_f32_16x16x32_bf16(
                false, af1, false, bf, (short)0, acc[1][t], false, false);
        }
        __syncthreads();
    }

    // Epilogue. D layout: VGPR j -> row = j + 8*hl, col = lane&15.
    #pragma unroll
    for (int s = 0; s < 2; ++s) {
        #pragma unroll
        for (int t = 0; t < 4; ++t) {
            const int col = n0 + 16 * t + l16;
            const float bv = (epi == EPI_SOFTPLUS) ? bias[col] : 0.f;
            #pragma unroll
            for (int j = 0; j < 8; ++j) {
                const int row = m0 + 32 * wave + 16 * s + j + 8 * hl;
                const size_t off = (size_t)row * ldc + col;
                float v = acc[s][t][j];
                if (epi == EPI_SOFTPLUS) {
                    v += bv;
                    v = (v > 20.f) ? v : log1pf(__expf(v));
                } else if (epi == EPI_RESID) {
                    v += resid[off];
                }
                C[off] = v;
            }
        }
    }
}

// ------------------------------ LayerNorm ----------------------------------
__global__ __launch_bounds__(128)
void layernorm_kernel(const float* __restrict__ X, float* __restrict__ U,
                      const float* __restrict__ w, const float* __restrict__ b)
{
    __shared__ float s1[128], s2[128];
    const int tok = blockIdx.x;
    const float* xr = X + (size_t)tok * D_MODEL;
    float v[4], sum = 0.f, sq = 0.f;
    #pragma unroll
    for (int i = 0; i < 4; ++i) {
        v[i] = xr[threadIdx.x + 128 * i];
        sum += v[i]; sq += v[i] * v[i];
    }
    s1[threadIdx.x] = sum; s2[threadIdx.x] = sq;
    __syncthreads();
    for (int s = 64; s > 0; s >>= 1) {
        if (threadIdx.x < s) {
            s1[threadIdx.x] += s1[threadIdx.x + s];
            s2[threadIdx.x] += s2[threadIdx.x + s];
        }
        __syncthreads();
    }
    const float mu  = s1[0] * (1.f / D_MODEL);
    const float var = s2[0] * (1.f / D_MODEL) - mu * mu;
    const float inv = rsqrtf(var + 1e-5f);
    float* ur = U + (size_t)tok * D_MODEL;
    #pragma unroll
    for (int i = 0; i < 4; ++i) {
        int c = threadIdx.x + 128 * i;
        ur[c] = (v[i] - mu) * inv * w[c] + b[c];
    }
}

// -------------------------- Causal conv + SiLU -----------------------------
__global__ __launch_bounds__(256)
void conv_silu_kernel(const float* __restrict__ XZ,     // [NTOK, 2*D_INNER]
                      const float* __restrict__ cw,     // [D_INNER, D_CONV]
                      const float* __restrict__ cb,     // [D_INNER]
                      float* __restrict__ XA)           // [NTOK, D_INNER]
{
    const int idx = blockIdx.x * blockDim.x + threadIdx.x;  // B*L*D_INNER
    const int d = idx & (D_INNER - 1);
    const int l = (idx >> 10) & (LSEQ - 1);
    const int b = idx >> 20;
    float acc = cb[d];
    const float* w4 = cw + d * D_CONV;
    #pragma unroll
    for (int k = 0; k < D_CONV; ++k) {
        const int ls = l - (D_CONV - 1) + k;
        if (ls >= 0)
            acc += XZ[((size_t)(b * LSEQ + ls)) * (2 * D_INNER) + d] * w4[k];
    }
    XA[idx] = acc * (1.f / (1.f + __expf(-acc)));   // silu
}

// ----------------------------- Selective scan ------------------------------
// One thread per (b, d) recurrence channel: 2048 independent chains.
__global__ __launch_bounds__(256)
void scan_kernel(const float* __restrict__ DEL,   // [NTOK, D_INNER] delta
                 const float* __restrict__ XA,    // [NTOK, D_INNER] conv+silu x
                 const float* __restrict__ DBL,   // [NTOK, 64] (dt|B|C)
                 const float* __restrict__ XZ,    // [NTOK, 2*D_INNER] (z gate)
                 const float* __restrict__ A_log, // [D_INNER, D_STATE]
                 const float* __restrict__ Dvec,  // [D_INNER]
                 float* __restrict__ Y)           // [NTOK, D_INNER]
{
    const int idx = blockIdx.x * blockDim.x + threadIdx.x;  // B*D_INNER
    const int d = idx & (D_INNER - 1);
    const int b = idx >> 10;

    float An[D_STATE], h[D_STATE];
    #pragma unroll
    for (int n = 0; n < D_STATE; ++n) {
        An[n] = -__expf(A_log[(size_t)d * D_STATE + n]);
        h[n] = 0.f;
    }
    const float Dd = Dvec[d];

    for (int l = 0; l < LSEQ; ++l) {
        const size_t t = (size_t)b * LSEQ + l;
        const float delta = DEL[t * D_INNER + d];
        const float xv    = XA[t * D_INNER + d];
        const float* Brow = DBL + t * 64 + DT_RANK;            // B: cols 32..47
        const float* Crow = DBL + t * 64 + DT_RANK + D_STATE;  // C: cols 48..63
        float y = 0.f;
        #pragma unroll
        for (int n = 0; n < D_STATE; ++n) {
            const float dA = __expf(delta * An[n]);
            h[n] = dA * h[n] + delta * Brow[n] * xv;
            y += h[n] * Crow[n];
        }
        y += xv * Dd;
        const float z = XZ[t * (2 * D_INNER) + D_INNER + d];
        y *= z * (1.f / (1.f + __expf(-z)));   // y * silu(z)
        Y[t * D_INNER + d] = y;
    }
}

// ------------------------- Concat / extract --------------------------------
__global__ __launch_bounds__(256)
void concat_kernel(const float* __restrict__ xc, const float* __restrict__ xt,
                   float* __restrict__ X)
{
    const int idx = blockIdx.x * blockDim.x + threadIdx.x;   // B*L*D_MODEL
    const int c = idx & (D_MODEL - 1);
    const int tok = idx >> 9;
    const int l = tok & (LSEQ - 1);
    const int b = tok >> 10;
    X[idx] = (l < NCTX)
        ? xc[((size_t)(b * NCTX + l)) * D_MODEL + c]
        : xt[((size_t)(b * NTGT + (l - NCTX))) * D_MODEL + c];
}

__global__ __launch_bounds__(256)
void extract_kernel(const float* __restrict__ X, float* __restrict__ out)
{
    const int idx = blockIdx.x * blockDim.x + threadIdx.x;   // B*NT*D_MODEL
    const int c = idx & (D_MODEL - 1);
    const int tok = idx >> 9;
    const int tt = tok & (NTGT - 1);
    const int b = tok >> 8;
    out[idx] = X[((size_t)(b * LSEQ + NCTX + tt)) * D_MODEL + c];
}

// ------------------------------ Launch -------------------------------------
extern "C" void kernel_launch(void* const* d_in, const int* in_sizes, int n_in,
                              void* d_out, int out_size, void* d_ws, size_t ws_size,
                              hipStream_t stream)
{
    const float* xc     = (const float*)d_in[0];
    const float* xt     = (const float*)d_in[1];
    const float* Win    = (const float*)d_in[2];
    const float* conv_w = (const float*)d_in[3];
    const float* conv_b = (const float*)d_in[4];
    const float* Wx     = (const float*)d_in[5];
    const float* Wdt    = (const float*)d_in[6];
    const float* b_dt   = (const float*)d_in[7];
    const float* A_log  = (const float*)d_in[8];
    const float* Dvec   = (const float*)d_in[9];
    const float* Wout   = (const float*)d_in[10];
    const float* ln_w   = (const float*)d_in[11];
    const float* ln_b   = (const float*)d_in[12];
    float* out = (float*)d_out;

    // Workspace layout (floats)
    float* ws  = (float*)d_ws;
    float* X   = ws;                                  // [NTOK, 512] residual
    float* UY  = X   + (size_t)NTOK * D_MODEL;        // ln out, later y
    float* XZ  = UY  + (size_t)NTOK * D_INNER;        // [NTOK, 2048]
    float* XA  = XZ  + (size_t)NTOK * 2 * D_INNER;    // [NTOK, 1024]
    float* DBL = XA  + (size_t)NTOK * D_INNER;        // [NTOK, 64]
    float* DEL = DBL + (size_t)NTOK * 64;             // [NTOK, 1024]

    concat_kernel<<<(NTOK * D_MODEL) / 256, 256, 0, stream>>>(xc, xt, X);

    for (int layer = 0; layer < NUM_LAYERS; ++layer) {
        const float* Win_l  = Win    + (size_t)layer * D_MODEL * 2 * D_INNER;
        const float* cw_l   = conv_w + (size_t)layer * D_INNER * D_CONV;
        const float* cb_l   = conv_b + (size_t)layer * D_INNER;
        const float* Wx_l   = Wx     + (size_t)layer * D_INNER * (DT_RANK + 2 * D_STATE);
        const float* Wdt_l  = Wdt    + (size_t)layer * DT_RANK * D_INNER;
        const float* bdt_l  = b_dt   + (size_t)layer * D_INNER;
        const float* Alog_l = A_log  + (size_t)layer * D_INNER * D_STATE;
        const float* Dv_l   = Dvec   + (size_t)layer * D_INNER;
        const float* Wout_l = Wout   + (size_t)layer * D_INNER * D_MODEL;
        const float* lnw_l  = ln_w   + (size_t)layer * D_MODEL;
        const float* lnb_l  = ln_b   + (size_t)layer * D_MODEL;

        // u = layer_norm(x)
        layernorm_kernel<<<NTOK, 128, 0, stream>>>(X, UY, lnw_l, lnb_l);

        // xz = u @ Win   (2048 x 512 x 2048)
        wmma_gemm_kernel<<<dim3((2 * D_INNER) / TILE_N, NTOK / TILE_M), 128, 0, stream>>>(
            UY, D_MODEL, Win_l, 2 * D_INNER, XZ, 2 * D_INNER,
            NTOK, 2 * D_INNER, D_MODEL, nullptr, nullptr, EPI_STORE);

        // x = silu(causal_conv(x) + conv_b)
        conv_silu_kernel<<<(NTOK * D_INNER) / 256, 256, 0, stream>>>(XZ, cw_l, cb_l, XA);

        // dbl = x @ Wx   (2048 x 1024 x 64)
        wmma_gemm_kernel<<<dim3(64 / TILE_N, NTOK / TILE_M), 128, 0, stream>>>(
            XA, D_INNER, Wx_l, 64, DBL, 64,
            NTOK, 64, D_INNER, nullptr, nullptr, EPI_STORE);

        // delta = softplus(dt @ Wdt + b_dt)  (2048 x 32 x 1024; dt = dbl[:, :32])
        wmma_gemm_kernel<<<dim3(D_INNER / TILE_N, NTOK / TILE_M), 128, 0, stream>>>(
            DBL, 64, Wdt_l, D_INNER, DEL, D_INNER,
            NTOK, D_INNER, DT_RANK, bdt_l, nullptr, EPI_SOFTPLUS);

        // selective scan + skip + gate: y (reuses UY)
        scan_kernel<<<(BATCH * D_INNER) / 256, 256, 0, stream>>>(
            DEL, XA, DBL, XZ, Alog_l, Dv_l, UY);

        // x = x + y @ Wout   (2048 x 1024 x 512)
        wmma_gemm_kernel<<<dim3(D_MODEL / TILE_N, NTOK / TILE_M), 128, 0, stream>>>(
            UY, D_INNER, Wout_l, D_MODEL, X, D_MODEL,
            NTOK, D_MODEL, D_INNER, nullptr, X, EPI_RESID);
    }

    // out = x[:, NC:, :]
    extract_kernel<<<(BATCH * NTGT * D_MODEL) / 256, 256, 0, stream>>>(X, out);
}
